// PatchGNNTransformer_73658689126730
// MI455X (gfx1250) — compile-verified, matrix operand order checked by reading, not run on hardware
//
#include <hip/hip_runtime.h>

// ---------------------------------------------------------------------------
// PatchGNNTransformer for MI455X (gfx1250, wave32, WMMA).
// fp32 residual stream; all GEMMs run f16 x f16 -> f32 via
// v_wmma_f32_16x16x32_f16, weights pre-converted to transposed f16
// (both operands K-contiguous). 128x128 block tile, 8 waves each 32x64,
// double-buffered LDS (one barrier per K-chunk, 8 WMMA per barrier).
// Edge-softmax via f32 atomics (qkv/x live in the 192MB L2).
// ---------------------------------------------------------------------------

typedef __attribute__((ext_vector_type(16))) _Float16 v16h;
typedef __attribute__((ext_vector_type(8)))  _Float16 v8h;
typedef __attribute__((ext_vector_type(8)))  float    v8f;

#define N_TOK   65536      // P*(COUNT+CLS)
#define HID     256
#define NHEAD   8
#define DHEAD   32
#define DEPTH   4
#define EXPH    1024       // EXP*HID
#define NPART   1024       // P
#define NCLS    4
#define SPOUT   256
#define NEDGE   262144
#define QKVW    768        // 3*HID

// ---------------------------------------------------------------------------
// Elementwise helpers
// ---------------------------------------------------------------------------
__global__ void copy_f32(const float* __restrict__ src, float* __restrict__ dst, int n) {
  int i = blockIdx.x * blockDim.x + threadIdx.x;
  if (i < n) dst[i] = src[i];
}

// W (K x Nc, row-major fp32)  ->  WT (Nc x K, row-major f16)
__global__ void convT(const float* __restrict__ W, _Float16* __restrict__ WT, int K, int Nc) {
  size_t idx = (size_t)blockIdx.x * blockDim.x + threadIdx.x;
  if (idx >= (size_t)K * Nc) return;
  int k = (int)(idx / Nc), n = (int)(idx % Nc);
  WT[(size_t)n * K + k] = (_Float16)W[idx];
}

// ---------------------------------------------------------------------------
// LayerNorm: one wave per 256-wide row, write f16
// ---------------------------------------------------------------------------
__global__ __launch_bounds__(256)
void ln_f16(const float* __restrict__ x, const float* __restrict__ g,
            const float* __restrict__ b, _Float16* __restrict__ out) {
  const int row  = blockIdx.x * 8 + (threadIdx.x >> 5);
  const int lane = threadIdx.x & 31;
  const float* xr = x + (size_t)row * HID;
  float4 v0 = ((const float4*)xr)[lane * 2];
  float4 v1 = ((const float4*)xr)[lane * 2 + 1];
  float vals[8] = {v0.x, v0.y, v0.z, v0.w, v1.x, v1.y, v1.z, v1.w};
  float s = 0.f, sq = 0.f;
#pragma unroll
  for (int j = 0; j < 8; ++j) { s += vals[j]; sq += vals[j] * vals[j]; }
#pragma unroll
  for (int m = 16; m >= 1; m >>= 1) {
    s  += __shfl_xor(s,  m, 32);
    sq += __shfl_xor(sq, m, 32);
  }
  const float mean = s * (1.f / HID);
  const float var  = sq * (1.f / HID) - mean * mean;
  const float inv  = rsqrtf(var + 1e-5f);
  v8h ov;
#pragma unroll
  for (int j = 0; j < 8; ++j) {
    const int c = lane * 8 + j;
    ov[j] = (_Float16)((vals[j] - mean) * inv * g[c] + b[c]);
  }
  *(v8h*)(out + (size_t)row * HID + lane * 8) = ov;
}

// ---------------------------------------------------------------------------
// WMMA GEMM: C(MxNc) = A(MxK,f16) * BT(NcxK,f16)^T  [+bias][+resid][relu]
// 128x128 block tile, 256 threads = 8 waves, each wave a 32x64 strip
// (2x4 = 8 wmma accumulators). Double-buffered LDS, 1 barrier per K-chunk.
// M, Nc multiples of 128; K multiple of 32 (true at every call site).
// ---------------------------------------------------------------------------
#define KB   32
#define LPAD 40   // LDS row pitch in halves (80B) -> conflict-free frag reads

__global__ __launch_bounds__(256)
void gemm_wmma(const _Float16* __restrict__ A, const _Float16* __restrict__ BT,
               const float* __restrict__ bias, const float* __restrict__ resid,
               float* __restrict__ outF, _Float16* __restrict__ outH,
               int M, int K, int Nc, int flags) {
  __shared__ _Float16 As[2][128 * LPAD];
  __shared__ _Float16 Bs[2][128 * LPAD];
  const int tid  = threadIdx.x;
  const int lane = tid & 31;
  const int wave = tid >> 5;
  const int wm   = wave & 3;    // 32-row group (0..3)
  const int wn   = wave >> 2;   // 64-col group (0..1)
  const int rowBase = blockIdx.y * 128;
  const int colBase = blockIdx.x * 128;

  v8f acc[2][4] = {};

  const int ldRow = tid >> 2;        // 0..63 (rows ldRow and ldRow+64)
  const int ldSeg = (tid & 3) * 8;   // 0,8,16,24 halves
  const _Float16* gA = A  + (size_t)(rowBase + ldRow) * K + ldSeg;
  const _Float16* gB = BT + (size_t)(colBase + ldRow) * K + ldSeg;
  const size_t rowStep = (size_t)64 * K;

  const int kh = (lane >> 4) * 8;    // A/B fragment K-half select
  const int fl = lane & 15;

  // fragment compute from LDS buffer `buf`
  auto compute = [&](int buf) {
    v16h af[2];
#pragma unroll
    for (int ti = 0; ti < 2; ++ti) {
      const int ar = wm * 32 + ti * 16 + fl;
      v8h lo = *(const v8h*)&As[buf][ar * LPAD + kh];
      v8h hi = *(const v8h*)&As[buf][ar * LPAD + kh + 16];
#pragma unroll
      for (int i = 0; i < 8; ++i) { af[ti][i] = lo[i]; af[ti][i + 8] = hi[i]; }
    }
#pragma unroll
    for (int tn = 0; tn < 4; ++tn) {
      const int bc = wn * 64 + tn * 16 + fl;
      v8h lo = *(const v8h*)&Bs[buf][bc * LPAD + kh];
      v8h hi = *(const v8h*)&Bs[buf][bc * LPAD + kh + 16];
      v16h bf;
#pragma unroll
      for (int i = 0; i < 8; ++i) { bf[i] = lo[i]; bf[i + 8] = hi[i]; }
#pragma unroll
      for (int ti = 0; ti < 2; ++ti)
        acc[ti][tn] = __builtin_amdgcn_wmma_f32_16x16x32_f16(
            false, af[ti], false, bf, (short)0, acc[ti][tn], false, false);
    }
  };

  // stage chunk 0 into buffer 0
  {
    *(v8h*)&As[0][ldRow * LPAD + ldSeg]        = *(const v8h*)(gA);
    *(v8h*)&As[0][(ldRow + 64) * LPAD + ldSeg] = *(const v8h*)(gA + rowStep);
    *(v8h*)&Bs[0][ldRow * LPAD + ldSeg]        = *(const v8h*)(gB);
    *(v8h*)&Bs[0][(ldRow + 64) * LPAD + ldSeg] = *(const v8h*)(gB + rowStep);
  }
  __syncthreads();

  int cur = 0;
  for (int k0 = KB; k0 < K; k0 += KB) {
    const int nxt = cur ^ 1;
    // issue global loads for next chunk (in flight during WMMA burst)
    v8h ra0 = *(const v8h*)(gA + k0);
    v8h ra1 = *(const v8h*)(gA + k0 + rowStep);
    v8h rb0 = *(const v8h*)(gB + k0);
    v8h rb1 = *(const v8h*)(gB + k0 + rowStep);
    if (k0 + KB < K) {  // near-cache prefetch of the chunk after next
      __builtin_prefetch(gA + k0 + KB, 0, 3);
      __builtin_prefetch(gB + k0 + KB, 0, 3);
    }
    // 8 WMMAs from current buffer
    compute(cur);
    // commit staged registers to the other buffer
    *(v8h*)&As[nxt][ldRow * LPAD + ldSeg]        = ra0;
    *(v8h*)&As[nxt][(ldRow + 64) * LPAD + ldSeg] = ra1;
    *(v8h*)&Bs[nxt][ldRow * LPAD + ldSeg]        = rb0;
    *(v8h*)&Bs[nxt][(ldRow + 64) * LPAD + ldSeg] = rb1;
    __syncthreads();
    cur = nxt;
  }
  compute(cur);

  // epilogue: C layout lane l, vgpr r -> m = r + 8*(l>=16), n = l&15
  const int mOff = (lane >> 4) * 8;
#pragma unroll
  for (int ti = 0; ti < 2; ++ti) {
#pragma unroll
    for (int tn = 0; tn < 4; ++tn) {
      const int col = colBase + wn * 64 + tn * 16 + fl;
      const float bia = bias ? bias[col] : 0.f;
#pragma unroll
      for (int r = 0; r < 8; ++r) {
        const int row = rowBase + wm * 32 + ti * 16 + mOff + r;
        const size_t o = (size_t)row * Nc + col;
        float v = acc[ti][tn][r] + bia;
        if (resid) v += resid[o];
        if (flags & 1) v = v > 0.f ? v : 0.f;
        if (outF) outF[o] = v;
        if (outH) outH[o] = (_Float16)v;
      }
    }
  }
}

// ---------------------------------------------------------------------------
// Edge attention (softmax over incoming edges per (dst, head))
// ---------------------------------------------------------------------------
__device__ inline void atomicMaxF(float* addr, float val) {
  int* ai = (int*)addr;
  int old = __float_as_int(*addr);
  while (__int_as_float(old) < val) {
    int assumed = old;
    old = atomicCAS(ai, assumed, __float_as_int(val));
    if (old == assumed) break;
  }
}

__global__ void attn_init(float* __restrict__ msg, float* __restrict__ m,
                          float* __restrict__ den) {
  int i = blockIdx.x * blockDim.x + threadIdx.x;
  if (i < N_TOK * HID) msg[i] = 0.f;
  if (i < N_TOK * NHEAD) { m[i] = -INFINITY; den[i] = 0.f; }
}

// score(e,h) = <q[dst,h,:], k[src,h,:]>/sqrt(DH); running max per (dst,h)
__global__ void attn_scores(const _Float16* __restrict__ qkv, const int* __restrict__ ei,
                            float* __restrict__ ebuf, float* __restrict__ mbuf) {
  int idx = blockIdx.x * blockDim.x + threadIdx.x;
  if (idx >= NEDGE * NHEAD) return;
  const int e = idx >> 3, h = idx & 7;
  const int src = ei[e], dst = ei[NEDGE + e];
  const v8h* qp = (const v8h*)(qkv + (size_t)dst * QKVW + h * DHEAD);
  const v8h* kp = (const v8h*)(qkv + (size_t)src * QKVW + HID + h * DHEAD);
  float s = 0.f;
#pragma unroll
  for (int j = 0; j < 4; ++j) {
    v8h q = qp[j], k = kp[j];
#pragma unroll
    for (int i = 0; i < 8; ++i) s += (float)q[i] * (float)k[i];
  }
  s *= 0.17677669529663687f;  // 1/sqrt(32)
  ebuf[idx] = s;
  atomicMaxF(&mbuf[dst * NHEAD + h], s);
}

// one block per edge: msg[dst,:] += exp(e-m) * v[src,:]; den[dst,h] += exp(e-m)
__global__ __launch_bounds__(256)
void attn_accum(const _Float16* __restrict__ qkv, const int* __restrict__ ei,
                const float* __restrict__ ebuf, const float* __restrict__ mbuf,
                float* __restrict__ msg, float* __restrict__ den) {
  const int e = blockIdx.x;
  const int c = threadIdx.x;       // h*32+d
  const int h = c >> 5;
  const int src = ei[e], dst = ei[NEDGE + e];
  const float w = __expf(ebuf[e * NHEAD + h] - mbuf[dst * NHEAD + h]);
  const float v = (float)qkv[(size_t)src * QKVW + 2 * HID + c];
  atomicAdd(&msg[(size_t)dst * HID + c], w * v);
  if ((c & 31) == 0) atomicAdd(&den[dst * NHEAD + h], w);
}

__global__ void attn_norm(const float* __restrict__ msg, const float* __restrict__ den,
                          _Float16* __restrict__ out) {
  int i = blockIdx.x * blockDim.x + threadIdx.x;
  if (i >= N_TOK * HID) return;
  const int n = i >> 8, h = (i & 255) >> 5;
  const float d = den[n * NHEAD + h];
  out[i] = (_Float16)(d > 0.f ? msg[i] / d : 0.f);
}

// ---------------------------------------------------------------------------
// Pooling head helpers
// ---------------------------------------------------------------------------
__global__ void scan_starts(const int* __restrict__ counts, int* __restrict__ starts) {
  __shared__ int s[NPART];
  const int t = threadIdx.x;
  const int v0 = counts[t] + NCLS;
  s[t] = v0;
  __syncthreads();
  for (int d = 1; d < NPART; d <<= 1) {
    int v = (t >= d) ? s[t - d] : 0;
    __syncthreads();
    s[t] += v;
    __syncthreads();
  }
  starts[t] = s[t] - v0;  // exclusive prefix
}

__global__ void gather_cls(const float* __restrict__ x, const int* __restrict__ starts,
                           _Float16* __restrict__ cls) {
  int i = blockIdx.x * blockDim.x + threadIdx.x;
  if (i >= NPART * NCLS * HID) return;
  const int p = i >> 10, rem = i & 1023;
  const int j = rem >> 8, c = rem & 255;
  cls[i] = (_Float16)x[(size_t)(starts[p] + j) * HID + c];
}

// ---------------------------------------------------------------------------
// Host side
// ---------------------------------------------------------------------------
extern "C" void kernel_launch(void* const* d_in, const int* in_sizes, int n_in,
                              void* d_out, int out_size, void* d_ws, size_t ws_size,
                              hipStream_t stream) {
  (void)in_sizes; (void)n_in; (void)out_size; (void)ws_size;
  const float* x_in  = (const float*)d_in[0];
  const int*   ei    = (const int*)d_in[1];
  const int*   pcnt  = (const int*)d_in[2];
  const float* ln1g  = (const float*)d_in[3];
  const float* ln1b  = (const float*)d_in[4];
  const float* Wqkv  = (const float*)d_in[5];
  const float* bqkv  = (const float*)d_in[6];
  const float* Wo    = (const float*)d_in[7];
  const float* bo    = (const float*)d_in[8];
  const float* ln2g  = (const float*)d_in[9];
  const float* ln2b  = (const float*)d_in[10];
  const float* W1    = (const float*)d_in[11];
  const float* b1    = (const float*)d_in[12];
  const float* W2    = (const float*)d_in[13];
  const float* b2    = (const float*)d_in[14];
  const float* spW1  = (const float*)d_in[15];
  const float* spb1  = (const float*)d_in[16];
  const float* spW2  = (const float*)d_in[17];
  const float* spb2  = (const float*)d_in[18];
  float* out = (float*)d_out;

  char* ws = (char*)d_ws;
  size_t off = 0;
  auto take = [&](size_t bytes) -> char* {
    char* p = ws + off;
    off += (bytes + 255) & ~(size_t)255;
    return p;
  };
  float*    xbuf  = (float*)   take((size_t)N_TOK * HID * 4);     //  64 MB residual stream
  _Float16* h16   = (_Float16*)take((size_t)N_TOK * HID * 2);     //  32 MB LN out / msg16
  _Float16* big16 = (_Float16*)take((size_t)N_TOK * EXPH * 2);    // 128 MB qkv / mlp hidden
  float*    msg   = (float*)   take((size_t)N_TOK * HID * 4);     //  64 MB attn accum
  float*    mbuf  = (float*)   take((size_t)N_TOK * NHEAD * 4);
  float*    den   = (float*)   take((size_t)N_TOK * NHEAD * 4);
  float*    ebuf  = (float*)   take((size_t)NEDGE * NHEAD * 4);
  _Float16* qkvT  = (_Float16*)take((size_t)DEPTH * QKVW * HID * 2);
  _Float16* woT   = (_Float16*)take((size_t)DEPTH * HID * HID * 2);
  _Float16* w1T   = (_Float16*)take((size_t)DEPTH * EXPH * HID * 2);
  _Float16* w2T   = (_Float16*)take((size_t)DEPTH * HID * EXPH * 2);
  _Float16* sp1T  = (_Float16*)take((size_t)EXPH * (NCLS * HID) * 2);
  _Float16* sp2T  = (_Float16*)take((size_t)SPOUT * EXPH * 2);
  int*      strt  = (int*)     take(NPART * 4);
  _Float16* clsb  = (_Float16*)take((size_t)NPART * NCLS * HID * 2);
  _Float16* clsh  = (_Float16*)take((size_t)NPART * EXPH * 2);

  const dim3 blk(256);
  auto eg = [](size_t n) { return dim3((unsigned)((n + 255) / 256)); };

  copy_f32<<<eg((size_t)N_TOK * HID), blk, 0, stream>>>(x_in, xbuf, N_TOK * HID);

  // one-time weight f16+transpose conversion (K-contiguous for WMMA B operand)
  for (int d = 0; d < DEPTH; ++d) {
    convT<<<eg((size_t)HID * QKVW), blk, 0, stream>>>(Wqkv + (size_t)d * HID * QKVW,
                                                      qkvT + (size_t)d * QKVW * HID, HID, QKVW);
    convT<<<eg((size_t)HID * HID),  blk, 0, stream>>>(Wo + (size_t)d * HID * HID,
                                                      woT + (size_t)d * HID * HID, HID, HID);
    convT<<<eg((size_t)HID * EXPH), blk, 0, stream>>>(W1 + (size_t)d * HID * EXPH,
                                                      w1T + (size_t)d * EXPH * HID, HID, EXPH);
    convT<<<eg((size_t)EXPH * HID), blk, 0, stream>>>(W2 + (size_t)d * EXPH * HID,
                                                      w2T + (size_t)d * HID * EXPH, EXPH, HID);
  }
  convT<<<eg((size_t)NCLS * HID * EXPH), blk, 0, stream>>>(spW1, sp1T, NCLS * HID, EXPH);
  convT<<<eg((size_t)EXPH * SPOUT),      blk, 0, stream>>>(spW2, sp2T, EXPH, SPOUT);

  for (int d = 0; d < DEPTH; ++d) {
    // h = LN1(x); qkv = h @ Wqkv + bqkv  (f16 out)
    ln_f16<<<dim3(N_TOK / 8), blk, 0, stream>>>(xbuf, ln1g + d * HID, ln1b + d * HID, h16);
    gemm_wmma<<<dim3(QKVW / 128, N_TOK / 128), blk, 0, stream>>>(
        h16, qkvT + (size_t)d * QKVW * HID, bqkv + d * QKVW,
        nullptr, nullptr, big16, N_TOK, HID, QKVW, 0);

    // edge softmax attention
    attn_init  <<<eg((size_t)N_TOK * HID), blk, 0, stream>>>(msg, mbuf, den);
    attn_scores<<<eg((size_t)NEDGE * NHEAD), blk, 0, stream>>>(big16, ei, ebuf, mbuf);
    attn_accum <<<dim3(NEDGE), blk, 0, stream>>>(big16, ei, ebuf, mbuf, msg, den);
    attn_norm  <<<eg((size_t)N_TOK * HID), blk, 0, stream>>>(msg, den, h16);

    // x = x + msg @ Wo + bo  (fused residual, fp32 out)
    gemm_wmma<<<dim3(HID / 128, N_TOK / 128), blk, 0, stream>>>(
        h16, woT + (size_t)d * HID * HID, bo + d * HID,
        xbuf, xbuf, nullptr, N_TOK, HID, HID, 0);

    // MLP: x = x + relu(LN2(x) @ W1 + b1) @ W2 + b2
    ln_f16<<<dim3(N_TOK / 8), blk, 0, stream>>>(xbuf, ln2g + d * HID, ln2b + d * HID, h16);
    gemm_wmma<<<dim3(EXPH / 128, N_TOK / 128), blk, 0, stream>>>(
        h16, w1T + (size_t)d * EXPH * HID, b1 + d * EXPH,
        nullptr, nullptr, big16, N_TOK, HID, EXPH, 1);
    gemm_wmma<<<dim3(HID / 128, N_TOK / 128), blk, 0, stream>>>(
        big16, w2T + (size_t)d * HID * EXPH, b2 + d * HID,
        xbuf, xbuf, nullptr, N_TOK, EXPH, HID, 0);
  }

  // pooling head
  scan_starts<<<dim3(1), dim3(NPART), 0, stream>>>(pcnt, strt);
  gather_cls<<<eg((size_t)NPART * NCLS * HID), blk, 0, stream>>>(xbuf, strt, clsb);
  gemm_wmma<<<dim3(EXPH / 128, NPART / 128), blk, 0, stream>>>(
      clsb, sp1T, spb1, nullptr, nullptr, clsh, NPART, NCLS * HID, EXPH, 1);
  gemm_wmma<<<dim3(SPOUT / 128, NPART / 128), blk, 0, stream>>>(
      clsh, sp2T, spb2, nullptr, out, nullptr, NPART, EXPH, SPOUT, 0);
}
